// SuccessiveSwinTransfomerBlock_86749749445281
// MI455X (gfx1250) — matrix operand order repaired and checked by reading, not compiled, not run on hardware
//
#include <hip/hip_runtime.h>

// ---------------------------------------------------------------------------
// Types for CDNA5 WMMA / TDM
// ---------------------------------------------------------------------------
typedef __attribute__((ext_vector_type(16))) __bf16 v16bf;
typedef __attribute__((ext_vector_type(8)))  float  v8f;
typedef __attribute__((ext_vector_type(4)))  unsigned int v4u;
typedef __attribute__((ext_vector_type(8)))  unsigned int v8u;

union FragBF { v16bf v; uint4 u[2]; };
union FragF  { v8f   v; float f[8]; };

__device__ __forceinline__ unsigned short f2bf(float f) {
  unsigned int u = __float_as_uint(f);
  unsigned int r = u + 0x7FFFu + ((u >> 16) & 1u);   // round-to-nearest-even
  return (unsigned short)(r >> 16);
}

#define WMMA_BF16(A, B, C) \
  __builtin_amdgcn_wmma_f32_16x16x32_bf16(false, (A), false, (B), (short)0, (C), false, false)

static __device__ const v8f kZeroAcc = {0.f,0.f,0.f,0.f,0.f,0.f,0.f,0.f};

// ---------------------------------------------------------------------------
// CDNA5 Tensor Data Mover: DMA a 2D tile (tileY rows x tileX elems of 2 bytes,
// row stride = strideElems) from global memory into LDS (packed, pitch=tileX).
// D# layout per CDNA5 ISA. Issued once per executing
// wave; tracked with TENSORcnt.
// ---------------------------------------------------------------------------
__device__ __forceinline__ void tdm_load_2d_b16(const void* gptr, void* lptr,
                                                unsigned tileX, unsigned tileY,
                                                unsigned strideElems) {
  unsigned long long ga = (unsigned long long)gptr;
  unsigned lds = (unsigned)(unsigned long long)lptr;   // LDS aperture low bits
  v4u g0;
  g0.x = 1u;                                  // count=1, user descriptor
  g0.y = lds;                                 // lds_addr[31:0]
  g0.z = (unsigned)ga;                        // global_addr[31:0]
  g0.w = ((unsigned)(ga >> 32) & 0x01FFFFFFu) // global_addr[56:32]
         | 0x80000000u;                       // type=2 ("image")
  v8u g1;
  g1[0] = 0x00010000u;                        // data_size=1 (2B); mask/pad/iter=0
  g1[1] = (tileX & 0xFFFFu) << 16;            // tensor_dim0[15:0] (== tile)
  g1[2] = (tileX >> 16) | ((tileY & 0xFFFFu) << 16);   // dim0 hi | dim1 lo
  g1[3] = (tileY >> 16) | (tileX << 16);      // dim1 hi | tile_dim0
  g1[4] = tileY & 0xFFFFu;                    // tile_dim1 (tile_dim2=0)
  g1[5] = strideElems;                        // tensor_dim0_stride[31:0]
  g1[6] = 0u;                                 // stride hi | dim1_stride lo
  g1[7] = 0u;
  asm volatile("tensor_load_to_lds %0, %1" :: "s"(g0), "s"(g1) : "memory");
}
__device__ __forceinline__ void wait_tensor0() {
  __builtin_amdgcn_s_wait_tensorcnt((short)0);
}

// ---------------------------------------------------------------------------
// fp32 (K x N) -> bf16 transposed (N x K); weights are small, done per launch
// ---------------------------------------------------------------------------
__global__ __launch_bounds__(256) void cvt_transpose_bf16_kernel(
    const float* __restrict__ in, unsigned short* __restrict__ out, int K, int N) {
  int i = blockIdx.x * 256 + threadIdx.x;
  if (i < K * N) {
    int k = i / N, n = i % N;
    out[(size_t)n * K + k] = f2bf(in[i]);
  }
}

// ---------------------------------------------------------------------------
// LayerNorm over D=768, one token per block, bf16 output
// ---------------------------------------------------------------------------
__global__ __launch_bounds__(256) void layernorm_bf16_kernel(const float* __restrict__ x,
                                                             const float* __restrict__ g,
                                                             const float* __restrict__ b,
                                                             unsigned short* __restrict__ out) {
  const int D = 768;
  int row = blockIdx.x;
  int t = threadIdx.x;
  const float* xr = x + (size_t)row * D;

  float v[3], s = 0.f, s2 = 0.f;
#pragma unroll
  for (int i = 0; i < 3; ++i) {
    v[i] = xr[t + i * 256];
    s += v[i];
    s2 += v[i] * v[i];
  }
  __shared__ float r1[256], r2[256];
  r1[t] = s; r2[t] = s2;
  __syncthreads();
  for (int o = 128; o > 0; o >>= 1) {
    if (t < o) { r1[t] += r1[t + o]; r2[t] += r2[t + o]; }
    __syncthreads();
  }
  float mu  = r1[0] * (1.f / D);
  float var = r2[0] * (1.f / D) - mu * mu;
  float inv = rsqrtf(var + 1e-5f);
  unsigned short* orow = out + (size_t)row * D;
#pragma unroll
  for (int i = 0; i < 3; ++i) {
    int c = t + i * 256;
    orow[c] = f2bf((v[i] - mu) * inv * g[c] + b[c]);
  }
}

// ---------------------------------------------------------------------------
// bf16 WMMA GEMM: C[M,N] = A[M,K] @ Wt[N,K]^T + bias, 128x128 tile, KSTEP=64.
// Double-buffered LDS fed by the Tensor Data Mover (one 2D-tile DMA per
// matrix per K-step; wave0 moves A, wave1 moves B; TENSORcnt + barrier sync).
//   EPI==0: out_bf16 = bf16(gemm + bias)
//   EPI==1: out_f32  = res_f32 + gemm + bias   (res may alias out: in-place)
// ---------------------------------------------------------------------------
template <int EPI>
__global__ __launch_bounds__(256) void gemm_bf16_kernel(
    const unsigned short* __restrict__ A,    // M x K bf16 row-major
    const unsigned short* __restrict__ Wt,   // N x K bf16 row-major (pre-transposed)
    const float* __restrict__ bias,          // N
    const float* __restrict__ res,           // EPI==1: M x N f32
    float* __restrict__ outf,                // EPI==1: M x N f32
    unsigned short* __restrict__ obf,        // EPI==0: M x N bf16
    int M, int N, int K) {
  __shared__ __align__(16) unsigned short AsBuf[2 * 128 * 64];   // [row][k]
  __shared__ __align__(16) unsigned short BtBuf[2 * 128 * 64];   // [n][k]

  const int t = threadIdx.x;
  const int lane = t & 31, wave = t >> 5;
  const int wm = wave >> 1, wn = wave & 1;          // 4 x 2 wave grid
  const int half = lane >> 4, ml = lane & 15;
  const int rowBase = blockIdx.x * 128;
  const int colBase = blockIdx.y * 128;

  v8f acc[2][4];
#pragma unroll
  for (int i = 0; i < 2; ++i)
#pragma unroll
    for (int j = 0; j < 4; ++j) acc[i][j] = kZeroAcc;

  // TDM-issue one 128x64 A tile + one 128x64 B tile into LDS buffer `buf`
  auto stage = [&](int k0, int buf) {
    if (wave == 0) {
      tdm_load_2d_b16(A + (size_t)rowBase * K + k0,
                      AsBuf + buf * (128 * 64), 64u, 128u, (unsigned)K);
    } else if (wave == 1) {
      tdm_load_2d_b16(Wt + (size_t)colBase * K + k0,
                      BtBuf + buf * (128 * 64), 64u, 128u, (unsigned)K);
    }
  };

  const int nk = K >> 6;                            // K/64 tiles
  stage(0, 0);
  wait_tensor0();
  __syncthreads();

  for (int kb = 0; kb < nk; ++kb) {
    if (kb + 1 < nk) stage((kb + 1) << 6, (kb + 1) & 1);   // overlap with compute

    const unsigned short* As = AsBuf + (kb & 1) * (128 * 64);
    const unsigned short* Bt = BtBuf + (kb & 1) * (128 * 64);
#pragma unroll
    for (int ks = 0; ks < 64; ks += 32) {
      FragBF a[2], bfr[4];
#pragma unroll
      for (int i = 0; i < 2; ++i) {
        int row = wm * 32 + i * 16 + ml;
        // A 16x32 bf16 frag: elems 0..15 -> K = ks + 16*half + e
        a[i].u[0] = *(const uint4*)(As + row * 64 + ks + 16 * half);
        a[i].u[1] = *(const uint4*)(As + row * 64 + ks + 16 * half + 8);
      }
#pragma unroll
      for (int j = 0; j < 4; ++j) {
        int n = wn * 64 + j * 16 + ml;
        // B 32x16 bf16 frag: lane column n, 16 contiguous K from ks+16*half
        bfr[j].u[0] = *(const uint4*)(Bt + n * 64 + ks + 16 * half);
        bfr[j].u[1] = *(const uint4*)(Bt + n * 64 + ks + 16 * half + 8);
      }
#pragma unroll
      for (int i = 0; i < 2; ++i)
#pragma unroll
        for (int j = 0; j < 4; ++j)
          acc[i][j] = WMMA_BF16(a[i].v, bfr[j].v, acc[i][j]);
    }
    wait_tensor0();                // waves 0/1 drain their TDM; others free
    __syncthreads();
  }

  // ---- epilogue ----
#pragma unroll
  for (int i = 0; i < 2; ++i)
#pragma unroll
    for (int j = 0; j < 4; ++j) {
      int col = colBase + wn * 64 + j * 16 + ml;
      float bv = bias[col];
      FragF c; c.v = acc[i][j];
#pragma unroll
      for (int v2 = 0; v2 < 8; ++v2) {
        int row = rowBase + wm * 32 + i * 16 + v2 + 8 * half;
        size_t off = (size_t)row * N + col;
        float val = c.f[v2] + bv;
        if (EPI == 0) obf[off] = f2bf(val);
        else          outf[off] = res[off] + val;
      }
    }
}

// ---------------------------------------------------------------------------
// Windowed attention, one (window, head) per 128-thread block.
// N = 49 tokens padded to 64, head_dim = 64. WMMA for Q·K^T and P·V.
// SHIFTED=1 folds the cyclic roll + additive shift-mask into indexing.
// ---------------------------------------------------------------------------
template <int SHIFTED>
__global__ __launch_bounds__(128) void attn_win_kernel(
    const unsigned short* __restrict__ qkv,  // Mtok x 2304 bf16
    unsigned short* __restrict__ aout) {     // Mtok x 768 bf16
  const int P = 56, WS = 7, WN = 8, SH = 3;
  __shared__ __align__(16) unsigned short Qs[64 * 64];  // [n][k]
  __shared__ __align__(16) unsigned short Ks[64 * 64];  // [n][k]
  __shared__ __align__(16) unsigned short Vt[64 * 64];  // [d][n] transposed
  __shared__ __align__(16) unsigned short Pb[64 * 64];  // softmax probs bf16
  __shared__ float S[64 * 64];
  __shared__ int tokIdx[64];
  __shared__ int maskIdx[64];

  const int win = blockIdx.x, head = blockIdx.y;
  const int b = win >> 6, wrem = win & 63;
  const int wy = wrem >> 3, wx = wrem & 7;
  const int t = threadIdx.x;

  // --- setup: token gather indices + shift-mask ids ---
  if (t < 64) {
    int n = t;
    if (n < WS * WS) {
      int i = n / WS, j = n % WS;
      int r = wy * WS + i, c = wx * WS + j;     // position in (rolled) image
      int rr = r, cc = c;
      if (SHIFTED) { rr = (r - SH + P) % P; cc = (c - SH + P) % P; }
      tokIdx[n] = b * (P * P) + rr * P + cc;
      int ri = (r + SH) % P, ci = (c + SH) % P; // roll(idx,(-s,-s)) lookup
      maskIdx[n] = ((ri + SH) / WS) * WN + ((ci + SH) / WS);
    } else {
      tokIdx[n] = -1;
      maskIdx[n] = -1;
    }
  }
  __syncthreads();

  // --- load Q, K (row-major) and V (transposed) into LDS; batch all global
  //     loads first so loadcnt waits are coalesced ---
  {
    int n = t >> 1, hh = t & 1;               // 64 rows x 2 half-rows
    int tok = tokIdx[n];
    const unsigned short* base =
        qkv + (tok >= 0 ? (size_t)tok * 2304 : 0) + head * 64 + hh * 32;
    uint4 z; z.x = z.y = z.z = z.w = 0u;
    uint4 qd[4], kd[4], vd[4];
    if (tok >= 0) {
#pragma unroll
      for (int i = 0; i < 4; ++i) {
        qd[i] = *(const uint4*)(base + i * 8);
        kd[i] = *(const uint4*)(base + 768 + i * 8);
        vd[i] = *(const uint4*)(base + 1536 + i * 8);
      }
    } else {
#pragma unroll
      for (int i = 0; i < 4; ++i) { qd[i] = z; kd[i] = z; vd[i] = z; }
    }
#pragma unroll
    for (int i = 0; i < 4; ++i) {
      *(uint4*)(Qs + n * 64 + hh * 32 + i * 8) = qd[i];
      *(uint4*)(Ks + n * 64 + hh * 32 + i * 8) = kd[i];
      const unsigned short* ve = (const unsigned short*)&vd[i];
#pragma unroll
      for (int e = 0; e < 8; ++e) Vt[(hh * 32 + i * 8 + e) * 64 + n] = ve[e];
    }
  }
  __syncthreads();

  const int lane = t & 31, wv = t >> 5;
  const int half = lane >> 4, ml = lane & 15;

  // --- S = Q @ K^T * scale : wave wv owns row-tile wv, 4 col-tiles ---
  {
    v8f accS[4];
#pragma unroll
    for (int j = 0; j < 4; ++j) accS[j] = kZeroAcc;
#pragma unroll
    for (int ks = 0; ks < 64; ks += 32) {
      FragBF a;
      int row = wv * 16 + ml;
      a.u[0] = *(const uint4*)(Qs + row * 64 + ks + 16 * half);
      a.u[1] = *(const uint4*)(Qs + row * 64 + ks + 16 * half + 8);
#pragma unroll
      for (int j = 0; j < 4; ++j) {
        FragBF bf;
        int col = j * 16 + ml;                 // B[k][n] = K[n][k] -> Ks row
        bf.u[0] = *(const uint4*)(Ks + col * 64 + ks + 16 * half);
        bf.u[1] = *(const uint4*)(Ks + col * 64 + ks + 16 * half + 8);
        accS[j] = WMMA_BF16(a.v, bf.v, accS[j]);
      }
    }
#pragma unroll
    for (int j = 0; j < 4; ++j) {
      FragF c; c.v = accS[j];
#pragma unroll
      for (int v2 = 0; v2 < 8; ++v2) {
        int row = wv * 16 + v2 + 8 * half;
        S[row * 64 + j * 16 + ml] = c.f[v2] * 0.125f;   // 1/sqrt(64)
      }
    }
  }
  __syncthreads();

  // --- softmax over valid 49 columns (+ faithful 1e-9 shift mask) ---
  if (t < 64) {
    int r = t;
    if (r < 49) {
      int mi = maskIdx[r];
      float m = -3.4e38f;
      for (int j = 0; j < 49; ++j) {
        float sv = S[r * 64 + j];
        if (SHIFTED && mi != maskIdx[j]) sv += 1e-9f;
        S[r * 64 + j] = sv;
        m = fmaxf(m, sv);
      }
      float sum = 0.f;
      for (int j = 0; j < 49; ++j) {
        float p = __expf(S[r * 64 + j] - m);
        S[r * 64 + j] = p;
        sum += p;
      }
      float inv = 1.f / sum;
      for (int j = 0; j < 49; ++j) Pb[r * 64 + j] = f2bf(S[r * 64 + j] * inv);
      for (int j = 49; j < 64; ++j) Pb[r * 64 + j] = 0;
    } else {
      for (int j = 0; j < 64; ++j) Pb[r * 64 + j] = 0;
    }
  }
  __syncthreads();

  // --- O = P @ V ; write gathered tokens (also undoes the roll) ---
  {
    v8f accO[4];
#pragma unroll
    for (int j = 0; j < 4; ++j) accO[j] = kZeroAcc;
#pragma unroll
    for (int ks = 0; ks < 64; ks += 32) {
      FragBF a;
      int row = wv * 16 + ml;
      a.u[0] = *(const uint4*)(Pb + row * 64 + ks + 16 * half);
      a.u[1] = *(const uint4*)(Pb + row * 64 + ks + 16 * half + 8);
#pragma unroll
      for (int j = 0; j < 4; ++j) {
        FragBF bf;
        int col = j * 16 + ml;                 // B[k][d] = V[k][d] -> Vt row d
        bf.u[0] = *(const uint4*)(Vt + col * 64 + ks + 16 * half);
        bf.u[1] = *(const uint4*)(Vt + col * 64 + ks + 16 * half + 8);
        accO[j] = WMMA_BF16(a.v, bf.v, accO[j]);
      }
    }
#pragma unroll
    for (int j = 0; j < 4; ++j) {
      FragF c; c.v = accO[j];
      int col = j * 16 + ml;
#pragma unroll
      for (int v2 = 0; v2 < 8; ++v2) {
        int row = wv * 16 + v2 + 8 * half;
        if (row < 49) {
          int tok = tokIdx[row];
          aout[(size_t)tok * 768 + head * 64 + col] = f2bf(c.f[v2]);
        }
      }
    }
  }
}

// ---------------------------------------------------------------------------
// Host launcher
// ---------------------------------------------------------------------------
extern "C" void kernel_launch(void* const* d_in, const int* in_sizes, int n_in,
                              void* d_out, int out_size, void* d_ws, size_t ws_size,
                              hipStream_t stream) {
  (void)in_sizes; (void)n_in; (void)out_size; (void)ws_size;
  const int M = 16 * 3136;       // 50176 tokens
  const int D = 768;
  const int N3 = 3 * D;          // 2304

  // inputs (setup_inputs order)
  const float* x0    = (const float*)d_in[0];
  const float* g1    = (const float*)d_in[1];
  const float* b1    = (const float*)d_in[2];
  const float* Wqkv1 = (const float*)d_in[3];
  const float* bqkv1 = (const float*)d_in[4];
  const float* Wo1   = (const float*)d_in[5];
  const float* bo1   = (const float*)d_in[6];
  const float* g2    = (const float*)d_in[7];
  const float* b2    = (const float*)d_in[8];
  const float* g3    = (const float*)d_in[9];
  const float* b3    = (const float*)d_in[10];
  const float* Wqkv2 = (const float*)d_in[11];
  const float* bqkv2 = (const float*)d_in[12];
  const float* Wo2   = (const float*)d_in[13];
  const float* bo2   = (const float*)d_in[14];
  const float* g4    = (const float*)d_in[15];
  const float* b4    = (const float*)d_in[16];
  const float* Wl    = (const float*)d_in[17];
  const float* bl    = (const float*)d_in[18];

  float* X = (float*)d_out;      // running residual (fp32), final answer

  // workspace carve-out
  char* w = (char*)d_ws;
  size_t off = 0;
  auto take = [&](size_t bytes) -> void* {
    void* p = w + off;
    off = (off + bytes + 255) & ~(size_t)255;
    return p;
  };
  unsigned short* Wqkv1t = (unsigned short*)take((size_t)D * N3 * 2);  // N3 x D
  unsigned short* Wo1t   = (unsigned short*)take((size_t)D * D * 2);   // D x D
  unsigned short* Wqkv2t = (unsigned short*)take((size_t)D * N3 * 2);
  unsigned short* Wo2t   = (unsigned short*)take((size_t)D * D * 2);
  unsigned short* Wlt    = (unsigned short*)take((size_t)D * D * 2);
  unsigned short* H      = (unsigned short*)take((size_t)M * D * 2);
  unsigned short* QKV    = (unsigned short*)take((size_t)M * N3 * 2);
  unsigned short* AO     = (unsigned short*)take((size_t)M * D * 2);

  // weight conversion + transpose to bf16 (Wt is N x K row-major)
  auto cvtT = [&](const float* src, unsigned short* dst, int K, int N) {
    int n = K * N;
    cvt_transpose_bf16_kernel<<<(n + 255) / 256, 256, 0, stream>>>(src, dst, K, N);
  };
  cvtT(Wqkv1, Wqkv1t, D, N3);
  cvtT(Wo1,   Wo1t,   D, D);
  cvtT(Wqkv2, Wqkv2t, D, N3);
  cvtT(Wo2,   Wo2t,   D, D);
  cvtT(Wl,    Wlt,    D, D);

  dim3 gQKV(M / 128, N3 / 128);
  dim3 gDD(M / 128, D / 128);
  dim3 gAttn(1024, 12);

  // ===== Block 1: W-MSA =====
  layernorm_bf16_kernel<<<M, 256, 0, stream>>>(x0, g1, b1, H);
  gemm_bf16_kernel<0><<<gQKV, 256, 0, stream>>>(H, Wqkv1t, bqkv1, nullptr, nullptr, QKV, M, N3, D);
  attn_win_kernel<0><<<gAttn, 128, 0, stream>>>(QKV, AO);
  gemm_bf16_kernel<1><<<gDD, 256, 0, stream>>>(AO, Wo1t, bo1, x0, X, nullptr, M, D, D);
  // MLP (shared linear)
  layernorm_bf16_kernel<<<M, 256, 0, stream>>>(X, g2, b2, H);
  gemm_bf16_kernel<1><<<gDD, 256, 0, stream>>>(H, Wlt, bl, X, X, nullptr, M, D, D);

  // ===== Block 2: SW-MSA =====
  layernorm_bf16_kernel<<<M, 256, 0, stream>>>(X, g3, b3, H);
  gemm_bf16_kernel<0><<<gQKV, 256, 0, stream>>>(H, Wqkv2t, bqkv2, nullptr, nullptr, QKV, M, N3, D);
  attn_win_kernel<1><<<gAttn, 128, 0, stream>>>(QKV, AO);
  gemm_bf16_kernel<1><<<gDD, 256, 0, stream>>>(AO, Wo2t, bo2, X, X, nullptr, M, D, D);
  // MLP (shared linear)
  layernorm_bf16_kernel<<<M, 256, 0, stream>>>(X, g4, b4, H);
  gemm_bf16_kernel<1><<<gDD, 256, 0, stream>>>(H, Wlt, bl, X, X, nullptr, M, D, D);
}
